// DechirpSTFT_28209345200285
// MI455X (gfx1250) — matrix operand order repaired and checked by reading, not compiled, not run on hardware
//
#include <hip/hip_runtime.h>
#include <hip/hip_bf16.h>
#include <math.h>

// ---------------------------------------------------------------------------
// Dechirp-STFT on gfx1250: per frame, resample + chirp, then 1024-pt FFT as
// two 32x32 complex matmuls (radix-32 Cooley-Tukey) on v_wmma_f32_16x16x32_f16.
// Trig tables (chirp, twiddle, DFT32) built cooperatively per block; the
// (d,k) warped-resample map is precomputed once per launch into d_ws.
// ---------------------------------------------------------------------------

typedef __attribute__((ext_vector_type(8)))  _Float16 v8h;
typedef __attribute__((ext_vector_type(16))) _Float16 v16h;
typedef __attribute__((ext_vector_type(8)))  float    v8f;

#define KFFT   1024
#define HOPSZ  512
#define PI2F   6.28318530717958647692f

// Shared: wframe(4096) + chirp f2(8192) + dftR(2048) + dftI(2048) + tw f2(8192)
#define SHARED_LDS_BYTES 24576
// Per-wave: ZtR(2048) + ZtI(2048) + outBuf(32*33*8 = 8448)
#define WAVE_LDS_BYTES   12544

__device__ inline v16h combine16(v8h lo, v8h hi) {
  v16h r;
#pragma unroll
  for (int h = 0; h < 8; ++h) { r[h] = lo[h]; r[h + 8] = hi[h]; }
  return r;
}

// A-fragment (16x32 f16) from row-major [32][32] f16 source, row block r0.
// ISA 7.12.2: lane<16 holds M=lane, K={0..7,16..23}; lane>=16 holds M=lane-16,
// K={8..15,24..31}. Each half is 8 contiguous halfwords -> ds_load_b128.
__device__ inline v16h loadA(const _Float16* s, int r0, int lane) {
  int m  = lane & 15;
  int kb = lane >> 4;
  const _Float16* row = s + (r0 + m) * 32;
  v8h lo = *(const v8h*)(row + 8 * kb);
  v8h hi = *(const v8h*)(row + 16 + 8 * kb);
  return combine16(lo, hi);
}

// B-fragment (32x16 f16), column block c0, reading the TRANSPOSE of B stored
// row-major [32][32]: s[n][k] == B[k][n]. Lanes 0-15: K={0..7,16..23} (N=lane),
// lanes 16-31: K={8..15,24..31} (N=lane-16).
__device__ inline v16h loadBt(const _Float16* s, int c0, int lane) {
  int nn = c0 + (lane & 15);
  int kb = (lane >> 4) * 8;
  const _Float16* col = s + nn * 32 + kb;
  v8h lo = *(const v8h*)(col);
  v8h hi = *(const v8h*)(col + 16);
  return combine16(lo, hi);
}

__device__ inline v16h vneg16(v16h x) {
  v16h r;
#pragma unroll
  for (int h = 0; h < 16; ++h) r[h] = -x[h];
  return r;
}

__device__ inline v8f wmma_f16(v16h a, v16h b, v8f c) {
  return __builtin_amdgcn_wmma_f32_16x16x32_f16(
      /*neg_a=*/false, a, /*neg_b=*/false, b,
      /*c_mod=*/(short)0, c, /*reuse_a=*/false, /*reuse_b=*/false);
}

// Apply twiddle (from LDS table) to a 16x16 C-tile; store row-major f16.
__device__ inline void twiddle_store(v8f tr, v8f ti, int r0, int c0, int lane,
                                     const float2* twT,
                                     _Float16* TpR, _Float16* TpI) {
  int n2 = c0 + (lane & 15);
  int mb = 8 * (lane >> 4);
#pragma unroll
  for (int r = 0; r < 8; ++r) {
    int k1 = r0 + r + mb;                       // C-layout: M = r + 8*(lane/16)
    float2 wv = twT[k1 * 32 + n2];              // (cos th, sin th), th<0 baked
    float xr = tr[r], xi = ti[r];
    TpR[k1 * 32 + n2] = (_Float16)(xr * wv.x - xi * wv.y);
    TpI[k1 * 32 + n2] = (_Float16)(xr * wv.y + xi * wv.x);
  }
}

__device__ inline void store_tile(v8f xr, v8f xi, int r0, int c0, int lane,
                                  float* oB) {
  int k2 = c0 + (lane & 15);
  int mb = 8 * (lane >> 4);
#pragma unroll
  for (int r = 0; r < 8; ++r) {
    int k1 = r0 + r + mb;
    float* p = oB + (k1 * 33 + k2) * 2;         // pad 33 -> <=2-way conflicts
    p[0] = xr[r];
    p[1] = xi[r];
  }
}

// --- setup: per-(d,k) warped-time resample map, constant across all frames --
__global__ void dechirp_tab_kernel(const float* __restrict__ dlnf,
                                   int* __restrict__ lo_tab,
                                   float* __restrict__ frac_tab) {
  int d = blockIdx.x;
  float beta = 2.0f * dlnf[d];
  bool  safe = fabsf(beta) < 1e-8f;
  float bs   = safe ? 1e-8f : beta;
  float ebm1 = __expf(bs) - 1.0f;
  float inv  = 2.0f / bs;
  for (int k = threadIdx.x; k < KFFT; k += blockDim.x) {
    float tau = (float)k * (1.0f / 1023.0f);
    float ts  = safe ? (-1.0f + 2.0f * tau)
                     : (inv * __logf(1.0f + tau * ebm1) - 1.0f);
    float pos = (ts + 1.0f) * 0.5f * 1023.0f;
    int lo = (int)pos;
    lo = lo < 0 ? 0 : (lo > 1022 ? 1022 : lo);
    lo_tab[d * KFFT + k]   = lo;
    frac_tab[d * KFFT + k] = pos - (float)lo;
  }
}

extern __shared__ char smem_raw[];

__global__ __launch_bounds__(256) void dechirp_stft_kernel(
    const float* __restrict__ x, const float* __restrict__ dlnf,
    const float* __restrict__ a_ptr, float* __restrict__ out,
    const int* __restrict__ lo_tab, const float* __restrict__ frac_tab,
    int nw, int n_sig, int nbatch, int use_tab) {
  float*    wframe = (float*)smem_raw;                  // 1024 f32
  float2*   chirpT = (float2*)(smem_raw + 4096);        // 1024 f2
  _Float16* dftR   = (_Float16*)(smem_raw + 12288);     // 1024 f16
  _Float16* dftI   = dftR + 1024;                       // 1024 f16
  float2*   twT    = (float2*)(smem_raw + 16384);       // 1024 f2
  char*     wbase  = smem_raw + SHARED_LDS_BYTES;

  const int tid  = threadIdx.x;
  const int lane = tid & 31;
  const int d    = tid >> 5;                            // wave id == trial id
  const int b    = blockIdx.x / nw;
  const int w    = blockIdx.x % nw;

  const float* frame_src = x + (size_t)b * n_sig + (size_t)w * HOPSZ;
  __builtin_prefetch(frame_src, 0, 3);                  // global_prefetch_b8

  // --- cooperative precompute: frame*hann, chirp, DFT32, twiddle ----------
  const float aa = a_ptr[0];
  for (int i = tid; i < KFFT; i += 256) {
    float win = 0.5f * (1.0f - __cosf(PI2F * (float)i * (1.0f / 1024.0f)));
    wframe[i] = frame_src[i] * win;

    float t = -1.0f + (float)i * (2.0f / 1023.0f);      // t_ident
    float cs, sn;
    __sincosf(aa * t * t, &sn, &cs);
    chirpT[i] = make_float2(cs, -sn);                   // e^{-i a t^2}

    int j = i >> 5, i2 = i & 31;
    __sincosf(PI2F * (float)(j * i2) * (1.0f / 32.0f), &sn, &cs);
    dftR[i] = (_Float16)cs;                             // e^{-2pi i jk/32}
    dftI[i] = (_Float16)(-sn);

    __sincosf((float)(j * i2) * (-PI2F / 1024.0f), &sn, &cs);
    twT[i] = make_float2(cs, sn);                       // W_1024^{k1*n2}
  }
  __syncthreads();

  // --- per-wave scratch ---------------------------------------------------
  _Float16* ZtR = (_Float16*)(wbase + d * WAVE_LDS_BYTES);  // Z^T [n2][n1]
  _Float16* ZtI = ZtR + 1024;
  float*    oB  = (float*)(ZtI + 1024);                     // 32x33 float2

  // --- warped-time resample + chirp, stored transposed for B-fragments ---
  if (use_tab) {
    const int*   lt = lo_tab + d * KFFT;
    const float* ft = frac_tab + d * KFFT;
#pragma unroll 4
    for (int n1 = 0; n1 < 32; ++n1) {
      int   k    = n1 * 32 + lane;                      // coalesced table read
      int   lo   = lt[k];
      float frac = ft[k];
      float res  = wframe[lo] * (1.0f - frac) + wframe[lo + 1] * frac;
      float2 ch  = chirpT[k];
      ZtR[lane * 32 + n1] = (_Float16)(res * ch.x);
      ZtI[lane * 32 + n1] = (_Float16)(res * ch.y);
    }
  } else {                                              // inline fallback
    float beta = 2.0f * dlnf[d];
    bool  safe = fabsf(beta) < 1e-8f;                   // wave-uniform
    float bs   = safe ? 1e-8f : beta;
    float ebm1 = __expf(bs) - 1.0f;
    float inv  = 2.0f / bs;
    for (int n1 = 0; n1 < 32; ++n1) {
      int   k   = n1 * 32 + lane;
      float tau = (float)k * (1.0f / 1023.0f);
      float ts  = safe ? (-1.0f + 2.0f * tau)
                       : (inv * __logf(1.0f + tau * ebm1) - 1.0f);
      float pos = (ts + 1.0f) * 0.5f * 1023.0f;
      int   lo  = (int)pos;
      lo = lo < 0 ? 0 : (lo > 1022 ? 1022 : lo);
      float frac = pos - (float)lo;
      float res  = wframe[lo] * (1.0f - frac) + wframe[lo + 1] * frac;
      float2 ch  = chirpT[k];
      ZtR[lane * 32 + n1] = (_Float16)(res * ch.x);
      ZtI[lane * 32 + n1] = (_Float16)(res * ch.y);
    }
  }

  // --- stage A: T = DFT32 x Z  (complex, 2x2 tiles, 16 WMMA) --------------
  v16h aR0 = loadA(dftR, 0, lane),  aR1 = loadA(dftR, 16, lane);
  v16h aI0 = loadA(dftI, 0, lane),  aI1 = loadA(dftI, 16, lane);
  v16h bR0 = loadBt(ZtR, 0, lane),  bR1 = loadBt(ZtR, 16, lane);
  v16h bI0 = loadBt(ZtI, 0, lane),  bI1 = loadBt(ZtI, 16, lane);
  v16h nI0 = vneg16(bI0),           nI1 = vneg16(bI1);

  v8f z8 = {};
  v8f tR00 = wmma_f16(aI0, nI0, wmma_f16(aR0, bR0, z8));
  v8f tI00 = wmma_f16(aI0, bR0, wmma_f16(aR0, bI0, z8));
  v8f tR01 = wmma_f16(aI0, nI1, wmma_f16(aR0, bR1, z8));
  v8f tI01 = wmma_f16(aI0, bR1, wmma_f16(aR0, bI1, z8));
  v8f tR10 = wmma_f16(aI1, nI0, wmma_f16(aR1, bR0, z8));
  v8f tI10 = wmma_f16(aI1, bR0, wmma_f16(aR1, bI0, z8));
  v8f tR11 = wmma_f16(aI1, nI1, wmma_f16(aR1, bR1, z8));
  v8f tI11 = wmma_f16(aI1, bR1, wmma_f16(aR1, bI1, z8));

  // --- twiddle W_1024^{k1 n2} from LDS table; store T' (reuse Zt) ---------
  twiddle_store(tR00, tI00, 0, 0, lane, twT, ZtR, ZtI);
  twiddle_store(tR01, tI01, 0, 16, lane, twT, ZtR, ZtI);
  twiddle_store(tR10, tI10, 16, 0, lane, twT, ZtR, ZtI);
  twiddle_store(tR11, tI11, 16, 16, lane, twT, ZtR, ZtI);

  // --- stage C: X2 = T' x DFT32 (DFT32 symmetric -> row reads for B) ------
  v16h cR0 = loadA(ZtR, 0, lane),   cR1 = loadA(ZtR, 16, lane);
  v16h cI0 = loadA(ZtI, 0, lane),   cI1 = loadA(ZtI, 16, lane);
  v16h dR0 = loadBt(dftR, 0, lane), dR1 = loadBt(dftR, 16, lane);
  v16h dI0 = loadBt(dftI, 0, lane), dI1 = loadBt(dftI, 16, lane);
  v16h mD0 = vneg16(dI0),           mD1 = vneg16(dI1);

  v8f xR00 = wmma_f16(cI0, mD0, wmma_f16(cR0, dR0, z8));
  v8f xI00 = wmma_f16(cI0, dR0, wmma_f16(cR0, dI0, z8));
  v8f xR01 = wmma_f16(cI0, mD1, wmma_f16(cR0, dR1, z8));
  v8f xI01 = wmma_f16(cI0, dR1, wmma_f16(cR0, dI1, z8));
  v8f xR10 = wmma_f16(cI1, mD0, wmma_f16(cR1, dR0, z8));
  v8f xI10 = wmma_f16(cI1, dR0, wmma_f16(cR1, dI0, z8));
  v8f xR11 = wmma_f16(cI1, mD1, wmma_f16(cR1, dR1, z8));
  v8f xI11 = wmma_f16(cI1, dR1, wmma_f16(cR1, dI1, z8));

  store_tile(xR00, xI00, 0, 0, lane, oB);
  store_tile(xR01, xI01, 0, 16, lane, oB);
  store_tile(xR10, xI10, 16, 0, lane, oB);
  store_tile(xR11, xI11, 16, 16, lane, oB);

  // --- X[k1 + 32*k2] = X2[k1][k2]; coalesced interleaved complex64 stores -
  size_t obase = (((size_t)d * nbatch + b) * (size_t)nw + w) * (size_t)KFFT;
  float2* og = (float2*)out + obase;
#pragma unroll 4
  for (int it = 0; it < 32; ++it) {
    const float* p = oB + (lane * 33 + it) * 2;         // k1 = lane, k2 = it
    og[it * 32 + lane] = make_float2(p[0], p[1]);
  }
}

extern "C" void kernel_launch(void* const* d_in, const int* in_sizes, int n_in,
                              void* d_out, int out_size, void* d_ws, size_t ws_size,
                              hipStream_t stream) {
  (void)n_in; (void)out_size;
  const float* x    = (const float*)d_in[0];
  const float* dlnf = (const float*)d_in[1];
  const float* a    = (const float*)d_in[2];

  const int D = in_sizes[1];                 // 8 trials -> 8 waves / block
  const int B = 2;                           // reference batch
  const int N = in_sizes[0] / B;             // 1<<20
  const int nw = (N - KFFT) / HOPSZ + 1;     // 2047

  const size_t ws_needed = (size_t)D * KFFT * 8;        // lo + frac tables
  const int use_tab = (ws_size >= ws_needed) ? 1 : 0;

  int*   lo_tab   = (int*)d_ws;
  float* frac_tab = (float*)((char*)d_ws + (size_t)D * KFFT * 4);

  if (use_tab)
    dechirp_tab_kernel<<<D, 256, 0, stream>>>(dlnf, lo_tab, frac_tab);

  const int blocks = B * nw;
  const size_t shbytes = SHARED_LDS_BYTES + (size_t)D * WAVE_LDS_BYTES;
  dechirp_stft_kernel<<<blocks, 32 * D, shbytes, stream>>>(
      x, dlnf, a, (float*)d_out, lo_tab, frac_tab, nw, N, B, use_tab);
}
